// CausalSelfAttention_13142599926097
// MI455X (gfx1250) — matrix operand order repaired and checked by reading, not compiled, not run on hardware
//
#include <hip/hip_runtime.h>
#include <stdint.h>

typedef __bf16 bf16;
typedef __attribute__((ext_vector_type(16))) __bf16 v16bf;
typedef __attribute__((ext_vector_type(8)))  __bf16 v8bf;
typedef __attribute__((ext_vector_type(8)))  float  v8f;

#define TSEQ 2048
#define DMODEL 1024
#define NHEAD 16
#define DHEAD 64

// Toggle for the CDNA5 async global->LDS path (ASYNCcnt-tracked DMA).
#define ATHENA_ASYNC 1

__device__ __forceinline__ v16bf combine16(v8bf a, v8bf b) {
    return __builtin_shufflevector(a, b, 0,1,2,3,4,5,6,7,8,9,10,11,12,13,14,15);
}

__device__ __forceinline__ v8f wmma_bf16(v16bf a, v16bf b, v8f c) {
    // D = A(16x32 bf16) * B(32x16 bf16) + C(16x16 f32)
    return __builtin_amdgcn_wmma_f32_16x16x32_bf16(
        /*neg_a=*/false, a, /*neg_b=*/false, b,
        /*c_mod=*/(short)0, c, /*reuse_a=*/false, /*reuse_b=*/false);
}

// A-fragment (16x32, bf16): two contiguous 8-elem chunks at k = 8*half and 16+8*half.
__device__ __forceinline__ v16bf load_afrag(const bf16* row, int half) {
    const bf16* p = row + half * 8;
    v8bf c0 = *(const v8bf*)(p);
    v8bf c1 = *(const v8bf*)(p + 16);
    return combine16(c0, c1);
}

// B-fragment (32x16, bf16): two contiguous 8-elem chunks at k = 8*half and 16+8*half.
__device__ __forceinline__ v16bf load_bfrag(const bf16* col, int half) {
    const bf16* p = col + half * 8;
    v8bf c0 = *(const v8bf*)(p);
    v8bf c1 = *(const v8bf*)(p + 16);
    return combine16(c0, c1);
}

// 16-byte async DMA global -> LDS (tracked on ASYNCcnt, bypasses VGPRs).
// LDS operand is the low 32 bits of the generic address (ISA 10.2).
__device__ __forceinline__ void async_cp16(const bf16* g, bf16* l) {
#if ATHENA_ASYNC
    asm volatile("global_load_async_to_lds_b128 %0, %1, off"
                 :: "v"((uint32_t)(uintptr_t)l), "v"((uint64_t)(uintptr_t)g)
                 : "memory");
#else
    *(v8bf*)l = *(const v8bf*)g;
#endif
}

__device__ __forceinline__ void async_wait0() {
#if ATHENA_ASYNC
    asm volatile("s_wait_asynccnt 0x0" ::: "memory");
#endif
}

// ---------------------------------------------------------------- convert f32 -> bf16
__global__ void cvt_bf16_kernel(const float* __restrict__ src, bf16* __restrict__ dst, int n) {
    int i = blockIdx.x * blockDim.x + threadIdx.x;
    if (i < n) dst[i] = (bf16)src[i];
}

// ---------------------------------------------------------------- GEMM: C = A @ W^T
// Block tile 128x128, 8 waves, wave = 32 rows x 64 cols (2 A-frags x 4 B-frags = 8 WMMA/k-step).
// Double-buffered LDS via XOR-toggled byte offset (uniform loop body -> no acc copies),
// async global->LDS staging overlapped with WMMA compute.
// mode 0: scatter bf16 into qkv [proj][B][H][T][DH]  (proj = blockIdx.z)
// mode 1: f32 row-major [M,1024]
__global__ __launch_bounds__(256) void gemm_wmma_kernel(
        const bf16* __restrict__ A, const bf16* __restrict__ Wbase,
        bf16* __restrict__ outq, float* __restrict__ outf, int mode) {
    const int mbase = blockIdx.x * 128;
    const int nbase = blockIdx.y * 128;
    const int proj  = blockIdx.z;
    const bf16* W = Wbase + (size_t)proj * (DMODEL * DMODEL);
    bf16* oq = outq ? outq + (size_t)proj * ((size_t)4 * TSEQ * DMODEL) : nullptr;

    const int tid  = threadIdx.x;
    const int w    = tid >> 5;
    const int l    = tid & 31;
    const int half = l >> 4;
    const int lm   = l & 15;
    const int wr   = w >> 1;   // 0..3 : row block of 32
    const int wc   = w & 1;    // 0..1 : col block of 64

    __shared__ __align__(16) bf16 At[2][128][40];  // stride 80B: 16B aligned, spreads banks
    __shared__ __align__(16) bf16 Wt[2][128][40];
    const uint32_t BUFB = sizeof(At[0]);           // 10240 bytes per buffer (same for Wt)

    bf16* const a_base = &At[0][0][0];
    bf16* const w_base = &Wt[0][0][0];

    const v8f z = (v8f){0.f,0.f,0.f,0.f,0.f,0.f,0.f,0.f};
    v8f c00 = z, c01 = z, c02 = z, c03 = z;   // row-block 0 (rows wr*32 + 0..15)
    v8f c10 = z, c11 = z, c12 = z, c13 = z;   // row-block 1 (rows wr*32 + 16..31)

    const int r0 = tid >> 2;   // 0..63
    const int cc = tid & 3;    // 16B chunk within 32-k slab

    // stage one 128x32 A slab + 128x32 W slab into the buffers at `a_dst`/`w_dst`
    auto stage = [&](bf16* a_dst, bf16* w_dst, int kt) {
        const int kof = kt * 32 + cc * 8;
        async_cp16(A + (size_t)(mbase + r0) * DMODEL + kof,      a_dst + r0 * 40 + cc * 8);
        async_cp16(A + (size_t)(mbase + r0 + 64) * DMODEL + kof, a_dst + (r0 + 64) * 40 + cc * 8);
        async_cp16(W + (size_t)(nbase + r0) * DMODEL + kof,      w_dst + r0 * 40 + cc * 8);
        async_cp16(W + (size_t)(nbase + r0 + 64) * DMODEL + kof, w_dst + (r0 + 64) * 40 + cc * 8);
    };

    stage(a_base, w_base, 0);
    async_wait0();
    __syncthreads();

    uint32_t tog = 0;
#pragma clang loop unroll(disable)
    for (int kt = 0; kt < DMODEL / 32; ++kt) {
        bf16* a_cur = (bf16*)((char*)a_base + tog);
        bf16* w_cur = (bf16*)((char*)w_base + tog);
        if (kt + 1 < DMODEL / 32) {       // DMA next tile during compute
            stage((bf16*)((char*)a_base + (tog ^ BUFB)),
                  (bf16*)((char*)w_base + (tog ^ BUFB)), kt + 1);
        }

        v16bf af0 = load_afrag(a_cur + (wr * 32 + lm) * 40, half);
        v16bf af1 = load_afrag(a_cur + (wr * 32 + 16 + lm) * 40, half);

        v16bf b0 = load_bfrag(w_cur + (wc * 64 + 0 * 16 + lm) * 40, half);
        c00 = wmma_bf16(af0, b0, c00);
        c10 = wmma_bf16(af1, b0, c10);
        v16bf b1 = load_bfrag(w_cur + (wc * 64 + 1 * 16 + lm) * 40, half);
        c01 = wmma_bf16(af0, b1, c01);
        c11 = wmma_bf16(af1, b1, c11);
        v16bf b2 = load_bfrag(w_cur + (wc * 64 + 2 * 16 + lm) * 40, half);
        c02 = wmma_bf16(af0, b2, c02);
        c12 = wmma_bf16(af1, b2, c12);
        v16bf b3 = load_bfrag(w_cur + (wc * 64 + 3 * 16 + lm) * 40, half);
        c03 = wmma_bf16(af0, b3, c03);
        c13 = wmma_bf16(af1, b3, c13);

        async_wait0();     // my DMA for next buffer landed
        __syncthreads();   // everyone's landed; reads of current buffer already drained
        tog ^= BUFB;
    }

    // epilogue (cold): repackage into arrays for uniform store loops
    v8f acc[2][4] = {{c00, c01, c02, c03}, {c10, c11, c12, c13}};
#pragma unroll
    for (int a = 0; a < 2; ++a) {
#pragma unroll
        for (int j = 0; j < 4; ++j) {
#pragma unroll
            for (int rr = 0; rr < 8; ++rr) {
                const int gm = mbase + wr * 32 + a * 16 + rr + 8 * half;
                const int gn = nbase + wc * 64 + j * 16 + lm;
                const float v = acc[a][j][rr];
                if (mode == 0) {
                    const int b = gm >> 11, t = gm & (TSEQ - 1);
                    const int hh = gn >> 6, d = gn & (DHEAD - 1);
                    oq[(((size_t)(b * NHEAD + hh) * TSEQ) + t) * DHEAD + d] = (bf16)v;
                } else {
                    outf[(size_t)gm * DMODEL + gn] = v;
                }
            }
        }
    }
}

// ---------------------------------------------------------------- flash attention
// grid: (T/64, H, B), block 128 (4 waves). Wave w owns query rows [qbase+16w, +16).
__global__ __launch_bounds__(128) void attn_wmma_kernel(
        const bf16* __restrict__ Q, const bf16* __restrict__ K,
        const bf16* __restrict__ V, bf16* __restrict__ Yatt) {
    const int qtile = blockIdx.x;
    const int h = blockIdx.y, b = blockIdx.z;
    const int tid  = threadIdx.x;
    const int w    = tid >> 5;
    const int l    = tid & 31;
    const int half = l >> 4;
    const int lm   = l & 15;
    const int qbase = qtile * 64;

    const size_t bh = (size_t)(b * NHEAD + h) * TSEQ * DHEAD;
    const bf16* Qp = Q + bh;
    const bf16* Kp = K + bh;
    const bf16* Vp = V + bh;

    __shared__ __align__(16) bf16  Kt[32][72];      // [key][dh], stride 144B
    __shared__ __align__(16) bf16  Vt[DHEAD][40];   // transposed: [dh][key], stride 80B
    __shared__ __align__(16) float Ss[4][16][33];   // per-wave S tile
    __shared__ __align__(16) bf16  Ps[4][16][40];   // per-wave P tile
    __shared__ float Al[4][16];                     // per-wave row scale

    // Preload Q fragments (16 rows x 64 dh = two 16x32 A-fragments)
    const int qrow = qbase + w * 16 + lm;
    v16bf aq[2];
#pragma unroll
    for (int kc = 0; kc < 2; ++kc)
        aq[kc] = load_afrag(Qp + (size_t)qrow * DHEAD + kc * 32, half);

    v8f o[4];
#pragma unroll
    for (int j = 0; j < 4; ++j) o[j] = (v8f){0.f,0.f,0.f,0.f,0.f,0.f,0.f,0.f};
    float m_run = -1e30f, l_run = 0.f;

    const int kt_end = qtile * 2 + 2;   // keys up to qbase+64 (mask trims the rest)
    for (int kt = 0; kt < kt_end; ++kt) {
        // stage K tile [32][64] (async DMA) and V^T tile [64][32] (manual transpose)
        {
            const int r = tid >> 2, cc = tid & 3;
            async_cp16(Kp + (size_t)(kt * 32 + r) * DHEAD + cc * 8, &Kt[r][cc * 8]);
            v8bf vchunk = *(const v8bf*)(Vp + (size_t)(kt * 32 + r) * DHEAD + cc * 8);
#pragma unroll
            for (int i = 0; i < 8; ++i) Vt[cc * 8 + i][r] = vchunk[i];
        }
        async_wait0();
        __syncthreads();

        // S = Q * K^T : 16 q-rows x 32 keys per wave
#pragma unroll
        for (int j = 0; j < 2; ++j) {
            v8f s = (v8f){0.f,0.f,0.f,0.f,0.f,0.f,0.f,0.f};
#pragma unroll
            for (int kc = 0; kc < 2; ++kc) {
                v16bf bf_ = load_bfrag(&Kt[j * 16 + lm][kc * 32], half);
                s = wmma_bf16(aq[kc], bf_, s);
            }
#pragma unroll
            for (int rr = 0; rr < 8; ++rr)
                Ss[w][rr + 8 * half][j * 16 + lm] = s[rr];
        }

        // online softmax: lanes 0..15 each own one row
        if (l < 16) {
            const int m  = l;
            const int qg = qbase + w * 16 + m;
            float sv[32];
            float mx = -1e30f;
#pragma unroll
            for (int c = 0; c < 32; ++c) {
                float s = Ss[w][m][c] * 0.125f;          // 1/sqrt(64)
                if (kt * 32 + c > qg) s = -1e30f;        // causal mask
                sv[c] = s;
                mx = fmaxf(mx, s);
            }
            const float m_new = fmaxf(m_run, mx);
            const float alpha = __expf(m_run - m_new);
            float rs = 0.f;
#pragma unroll
            for (int c = 0; c < 32; ++c) {
                const float p = __expf(sv[c] - m_new);
                rs += p;
                Ps[w][m][c] = (bf16)p;
            }
            l_run = l_run * alpha + rs;
            m_run = m_new;
            Al[w][m] = alpha;
        }
        __syncthreads();

        // rescale O accumulators, then O += P * V
#pragma unroll
        for (int j = 0; j < 4; ++j)
#pragma unroll
            for (int rr = 0; rr < 8; ++rr)
                o[j][rr] *= Al[w][rr + 8 * half];

        v16bf ap = load_afrag(&Ps[w][lm][0], half);
#pragma unroll
        for (int j = 0; j < 4; ++j) {
            v16bf bf_ = load_bfrag(&Vt[j * 16 + lm][0], half);
            o[j] = wmma_bf16(ap, bf_, o[j]);
        }
        __syncthreads();
    }

    if (l < 16) Al[w][l] = 1.0f / l_run;
    __syncthreads();

#pragma unroll
    for (int j = 0; j < 4; ++j) {
#pragma unroll
        for (int rr = 0; rr < 8; ++rr) {
            const int row = qbase + w * 16 + rr + 8 * half;
            const int col = h * DHEAD + j * 16 + lm;
            const float v = o[j][rr] * Al[w][rr + 8 * half];
            Yatt[(size_t)(b * TSEQ + row) * DMODEL + col] = (bf16)v;
        }
    }
}

// ---------------------------------------------------------------- launch
extern "C" void kernel_launch(void* const* d_in, const int* in_sizes, int n_in,
                              void* d_out, int out_size, void* d_ws, size_t ws_size,
                              hipStream_t stream) {
    const float* x  = (const float*)d_in[0];
    const float* Wq = (const float*)d_in[1];
    const float* Wk = (const float*)d_in[2];
    const float* Wv = (const float*)d_in[3];
    const float* Wo = (const float*)d_in[4];
    float* out = (float*)d_out;

    const size_t M = (size_t)4 * TSEQ;                 // 8192
    char* ws = (char*)d_ws;
    size_t off = 0;
    bf16* xbf  = (bf16*)(ws + off); off += M * DMODEL * sizeof(bf16);            // 16 MiB
    bf16* wbf  = (bf16*)(ws + off); off += 4ull * DMODEL * DMODEL * sizeof(bf16);// 8 MiB
    bf16* qkv  = (bf16*)(ws + off); off += 3ull * M * DMODEL * sizeof(bf16);     // 48 MiB
    bf16* yatt = (bf16*)(ws + off); off += M * DMODEL * sizeof(bf16);            // 16 MiB

    // 1) downcast activations + weights to bf16
    {
        const int nx = (int)(M * DMODEL);
        cvt_bf16_kernel<<<(nx + 255) / 256, 256, 0, stream>>>(x, xbf, nx);
        const int nw = DMODEL * DMODEL;
        cvt_bf16_kernel<<<(nw + 255) / 256, 256, 0, stream>>>(Wq, wbf + 0ull * nw, nw);
        cvt_bf16_kernel<<<(nw + 255) / 256, 256, 0, stream>>>(Wk, wbf + 1ull * nw, nw);
        cvt_bf16_kernel<<<(nw + 255) / 256, 256, 0, stream>>>(Wv, wbf + 2ull * nw, nw);
        cvt_bf16_kernel<<<(nw + 255) / 256, 256, 0, stream>>>(Wo, wbf + 3ull * nw, nw);
    }

    // 2) Q,K,V projections (grid.z = projection index into wbf / qkv)
    gemm_wmma_kernel<<<dim3((unsigned)(M / 128), DMODEL / 128, 3), 256, 0, stream>>>(
        xbf, wbf, qkv, nullptr, /*mode=*/0);

    // 3) causal flash attention
    attn_wmma_kernel<<<dim3(TSEQ / 64, NHEAD, 4), 128, 0, stream>>>(
        qkv,
        qkv + 1ull * M * DMODEL,
        qkv + 2ull * M * DMODEL,
        yatt);

    // 4) output projection -> f32 d_out
    gemm_wmma_kernel<<<dim3((unsigned)(M / 128), DMODEL / 128, 1), 256, 0, stream>>>(
        yatt, wbf + 3ull * DMODEL * DMODEL, nullptr, out, /*mode=*/1);

    (void)in_sizes; (void)n_in; (void)out_size; (void)ws_size;
}
